// MultiLayer_BTreeLSTM_60421599920141
// MI455X (gfx1250) — matrix operand order repaired and checked by reading, not compiled, not run on hardware
//
#include <hip/hip_runtime.h>
#include <hip/hip_bf16.h>

// ---------------------------------------------------------------------------
// MultiLayer bidirectional TreeLSTM for MI455X (gfx1250, wave32, WMMA bf16)
// N=1024 nodes, D=1024 feat, H=512 hidden, L=2 layers, complete binary tree.
// Tree recurrence is level-parallel: 11 levels, batched WMMA GEMM per level.
// GEMM: 32x64 wave tile, K-loop unrolled x2 with even/odd register sets
// (no ping-pong copies; loads of one set overlap WMMAs of the other).
// ---------------------------------------------------------------------------

typedef __attribute__((ext_vector_type(16))) __bf16 v16bf;
typedef __attribute__((ext_vector_type(8)))  __bf16 v8bf;
typedef __attribute__((ext_vector_type(8)))  float  v8f;

#define NNODE 1024
#define DDIM  1024
#define HDIM  512
#define SIXH  3072
#define FIVEH 2560
#define LNUM  2

__device__ __forceinline__ float sigmoidf_(float x) { return 1.0f / (1.0f + __expf(-x)); }

__device__ __forceinline__ v16bf loadpair(const __bf16* p0, const __bf16* p1) {
  v8bf lo = *reinterpret_cast<const v8bf*>(p0);
  v8bf hi = *reinterpret_cast<const v8bf*>(p1);
  v16bf r;
#pragma unroll
  for (int i = 0; i < 8; ++i) { r[i] = lo[i]; r[i + 8] = hi[i]; }
  return r;
}

__device__ __forceinline__ v16bf zero16() {
  v16bf r;
#pragma unroll
  for (int i = 0; i < 16; ++i) r[i] = (__bf16)0.0f;
  return r;
}

__device__ __forceinline__ v8f wmma_bf16(v16bf a, v16bf b, v8f c) {
  return __builtin_amdgcn_wmma_f32_16x16x32_bf16(false, a, false, b, (short)0, c, false, false);
}

// ---------------------------------------------------------------------------
// Elementwise helpers
// ---------------------------------------------------------------------------
__global__ void f32_to_bf16_kernel(const float* __restrict__ src, __bf16* __restrict__ dst, int n) {
  int i = blockIdx.x * blockDim.x + threadIdx.x;
  int stride = gridDim.x * blockDim.x;
  for (; i < n; i += stride) dst[i] = (__bf16)src[i];
}

__global__ void combine_bias3_kernel(const float* __restrict__ a, const float* __restrict__ b,
                                     const float* __restrict__ c, float* __restrict__ o, int n) {
  int i = blockIdx.x * blockDim.x + threadIdx.x;
  if (i < n) o[i] = a[i] + b[i] + c[i];
}

__global__ void combine_bias2_kernel(const float* __restrict__ a, const float* __restrict__ b,
                                     float* __restrict__ o, int n) {
  int i = blockIdx.x * blockDim.x + threadIdx.x;
  if (i < n) o[i] = a[i] + b[i];
}

// ---------------------------------------------------------------------------
// GEMM: C[M,Ncols] = A[M,K](bf16,rm) @ W[Ncols,K]^T(bf16,rm) + bias
// WG = 256 threads = 8 waves; wave tile 32x64 (2x4 accum frags, B reused
// across both row halves); WG tile 64x256. grid = (Ncols/256, M/64).
// Requires K % 64 == 0 and K >= 64 (all callers use K = 1024).
// ---------------------------------------------------------------------------
#define GEMM_LOAD(SUF, KK)                                                        \
  a0##SUF = loadpair(arow0 + (KK) + half * 16, arow0 + (KK) + half * 16 + 8);     \
  a1##SUF = loadpair(arow1 + (KK) + half * 16, arow1 + (KK) + half * 16 + 8);     \
  b0##SUF = loadpair(wrow0 + (KK) + half * 16, wrow0 + (KK) + half * 16 + 8);     \
  b1##SUF = loadpair(wrow1 + (KK) + half * 16, wrow1 + (KK) + half * 16 + 8);     \
  b2##SUF = loadpair(wrow2 + (KK) + half * 16, wrow2 + (KK) + half * 16 + 8);     \
  b3##SUF = loadpair(wrow3 + (KK) + half * 16, wrow3 + (KK) + half * 16 + 8)

#define GEMM_MMA(SUF)                                       \
  acc[0][0] = wmma_bf16(a0##SUF, b0##SUF, acc[0][0]);       \
  acc[1][0] = wmma_bf16(a1##SUF, b0##SUF, acc[1][0]);       \
  acc[0][1] = wmma_bf16(a0##SUF, b1##SUF, acc[0][1]);       \
  acc[1][1] = wmma_bf16(a1##SUF, b1##SUF, acc[1][1]);       \
  acc[0][2] = wmma_bf16(a0##SUF, b2##SUF, acc[0][2]);       \
  acc[1][2] = wmma_bf16(a1##SUF, b2##SUF, acc[1][2]);       \
  acc[0][3] = wmma_bf16(a0##SUF, b3##SUF, acc[0][3]);       \
  acc[1][3] = wmma_bf16(a1##SUF, b3##SUF, acc[1][3])

__global__ __launch_bounds__(256) void gemm_bf16_wmma(
    const __bf16* __restrict__ A, int lda,
    const __bf16* __restrict__ W, int ldw,
    const float* __restrict__ bias,
    float* __restrict__ C, int ldc, int K) {
  const int lane = threadIdx.x & 31;
  const int wave = threadIdx.x >> 5;
  const int rowBase = blockIdx.y * 64 + (wave >> 2) * 32;
  const int colBase = blockIdx.x * 256 + (wave & 3) * 64;
  const int r = lane & 15, half = lane >> 4;

  const __bf16* arow0 = A + (size_t)(rowBase + r) * lda;
  const __bf16* arow1 = A + (size_t)(rowBase + 16 + r) * lda;
  const __bf16* wrow0 = W + (size_t)(colBase + 0 * 16 + r) * ldw;
  const __bf16* wrow1 = W + (size_t)(colBase + 1 * 16 + r) * ldw;
  const __bf16* wrow2 = W + (size_t)(colBase + 2 * 16 + r) * ldw;
  const __bf16* wrow3 = W + (size_t)(colBase + 3 * 16 + r) * ldw;

  v8f acc[2][4];
#pragma unroll
  for (int h2 = 0; h2 < 2; ++h2)
#pragma unroll
    for (int f = 0; f < 4; ++f)
#pragma unroll
      for (int e = 0; e < 8; ++e) acc[h2][f][e] = 0.0f;

  v16bf a0E, a1E, b0E, b1E, b2E, b3E;  // even K-step fragments
  v16bf a0O, a1O, b0O, b1O, b2O, b3O;  // odd  K-step fragments

  GEMM_LOAD(E, 0);
  int k = 0;
  for (; k + 64 < K; k += 64) {
    GEMM_LOAD(O, k + 32);
    GEMM_MMA(E);
    GEMM_LOAD(E, k + 64);
    GEMM_MMA(O);
  }
  GEMM_LOAD(O, k + 32);
  GEMM_MMA(E);
  GEMM_MMA(O);

  const int m0 = half * 8;
#pragma unroll
  for (int h2 = 0; h2 < 2; ++h2) {
#pragma unroll
    for (int f = 0; f < 4; ++f) {
      const int col = colBase + f * 16 + r;
      const float bv = bias[col];
#pragma unroll
      for (int e = 0; e < 8; ++e) {
        C[(size_t)(rowBase + h2 * 16 + m0 + e) * ldc + col] = acc[h2][f][e] + bv;
      }
    }
  }
}

// ---------------------------------------------------------------------------
// Forward (leaves->root) level kernel.
// Wave owns 16 nodes x 16 hidden cols; 6 gate-block accumulators in regs.
// g = pre_x[node] + h[left]@Wl^T + h[right]@Wr^T   (biases folded in pre_x)
// All 12 B fragments of a K-step load before the WMMA chain (partial waits).
// grid.x = mtiles*4, block 256 (8 waves -> 8 j-chunks per WG, 32 total).
// ---------------------------------------------------------------------------
__global__ __launch_bounds__(256) void fwd_level_kernel(
    const __bf16* __restrict__ hsrc,   // ld DDIM, hf at col 0
    const int* __restrict__ left, const int* __restrict__ right,
    const float* __restrict__ pre_x,   // [N, 6H]
    const float* __restrict__ pre_px,  // [N, H]
    const __bf16* __restrict__ Wl,     // [6H, H] bf16
    const __bf16* __restrict__ Wr,     // [6H, H] bf16
    float* __restrict__ cbuf,          // [N, H]
    float* __restrict__ hout_f32,      // ld DDIM (already col-offset)
    __bf16* __restrict__ hout_bf,      // ld DDIM (already col-offset)
    int lo, int count) {
  const int lane = threadIdx.x & 31;
  const int wave = threadIdx.x >> 5;
  const int mtile = blockIdx.x >> 2;
  const int j0 = ((blockIdx.x & 3) * 8 + wave) * 16;
  const int r = lane & 15, half = lane >> 4;

  const int anode = lo + mtile * 16 + r;
  const bool avalid = (mtile * 16 + r) < count;
  const int li = avalid ? left[anode] : -1;
  const int ri = avalid ? right[anode] : -1;
  const __bf16* lrow = (li >= 0) ? hsrc + (size_t)li * DDIM : nullptr;
  const __bf16* rrow = (ri >= 0) ? hsrc + (size_t)ri * DDIM : nullptr;

  v8f acc[6];
#pragma unroll
  for (int g = 0; g < 6; ++g)
#pragma unroll
    for (int e = 0; e < 8; ++e) acc[g][e] = 0.0f;

  for (int k = 0; k < HDIM; k += 32) {
    v16bf al = lrow ? loadpair(lrow + k + half * 16, lrow + k + half * 16 + 8) : zero16();
    v16bf ar = rrow ? loadpair(rrow + k + half * 16, rrow + k + half * 16 + 8) : zero16();
    v16bf bl[6], br[6];
#pragma unroll
    for (int g = 0; g < 6; ++g) {
      const __bf16* wl = Wl + (size_t)(g * HDIM + j0 + r) * HDIM + k + half * 16;
      bl[g] = loadpair(wl, wl + 8);
    }
#pragma unroll
    for (int g = 0; g < 6; ++g) {
      const __bf16* wr = Wr + (size_t)(g * HDIM + j0 + r) * HDIM + k + half * 16;
      br[g] = loadpair(wr, wr + 8);
    }
#pragma unroll
    for (int g = 0; g < 6; ++g) acc[g] = wmma_bf16(al, bl[g], acc[g]);
#pragma unroll
    for (int g = 0; g < 6; ++g) acc[g] = wmma_bf16(ar, br[g], acc[g]);
  }

  const int j = j0 + r;  // output hidden column
#pragma unroll
  for (int e = 0; e < 8; ++e) {
    const int m = half * 8 + e;
    if (mtile * 16 + m >= count) continue;
    const int node = lo + mtile * 16 + m;
    const size_t gb = (size_t)node * SIXH + j;
    const float g0 = acc[0][e] + pre_x[gb + 0 * HDIM];
    const float g1 = acc[1][e] + pre_x[gb + 1 * HDIM];
    const float g2 = acc[2][e] + pre_x[gb + 2 * HDIM];
    const float g3 = acc[3][e] + pre_x[gb + 3 * HDIM];
    const float g4 = acc[4][e] + pre_x[gb + 4 * HDIM];
    const float g5 = acc[5][e] + pre_x[gb + 5 * HDIM];
    const float ig = sigmoidf_(g0), og = sigmoidf_(g1);
    const float fl = sigmoidf_(g2), fr = sigmoidf_(g3);
    const float u = tanhf(g4), rg = sigmoidf_(g5);
    const int lie = left[node], rie = right[node];
    const float lc = (lie >= 0) ? cbuf[(size_t)lie * HDIM + j] : 0.0f;
    const float rc = (rie >= 0) ? cbuf[(size_t)rie * HDIM + j] : 0.0f;
    const float c = ig * u + fl * lc + fr * rc;
    const float hh = og * tanhf(c);
    const float hf = rg * hh + (1.0f - rg) * pre_px[(size_t)node * HDIM + j];
    cbuf[(size_t)node * HDIM + j] = c;
    hout_f32[(size_t)node * DDIM + j] = hf;
    hout_bf[(size_t)node * DDIM + j] = (__bf16)hf;
  }
}

// ---------------------------------------------------------------------------
// Backward (root->leaves) level kernel: 5 gates, single parent hidden.
// ---------------------------------------------------------------------------
__global__ __launch_bounds__(256) void bwd_level_kernel(
    const __bf16* __restrict__ hsrc,   // ld DDIM, hb (already col-offset)
    const int* __restrict__ parent,
    const float* __restrict__ pre_x,   // [N, 5H]
    const float* __restrict__ pre_px,  // [N, H]
    const __bf16* __restrict__ Wh,     // [5H, H] bf16
    float* __restrict__ cbuf,          // [N, H]
    float* __restrict__ hout_f32,      // ld DDIM (already col-offset)
    __bf16* __restrict__ hout_bf,      // ld DDIM (already col-offset)
    int lo, int count) {
  const int lane = threadIdx.x & 31;
  const int wave = threadIdx.x >> 5;
  const int mtile = blockIdx.x >> 2;
  const int j0 = ((blockIdx.x & 3) * 8 + wave) * 16;
  const int r = lane & 15, half = lane >> 4;

  const int anode = lo + mtile * 16 + r;
  const bool avalid = (mtile * 16 + r) < count;
  const int pi = avalid ? parent[anode] : -1;
  const __bf16* prow = (pi >= 0) ? hsrc + (size_t)pi * DDIM : nullptr;

  v8f acc[5];
#pragma unroll
  for (int g = 0; g < 5; ++g)
#pragma unroll
    for (int e = 0; e < 8; ++e) acc[g][e] = 0.0f;

  for (int k = 0; k < HDIM; k += 32) {
    v16bf a = prow ? loadpair(prow + k + half * 16, prow + k + half * 16 + 8) : zero16();
    v16bf b[5];
#pragma unroll
    for (int g = 0; g < 5; ++g) {
      const __bf16* wp = Wh + (size_t)(g * HDIM + j0 + r) * HDIM + k + half * 16;
      b[g] = loadpair(wp, wp + 8);
    }
#pragma unroll
    for (int g = 0; g < 5; ++g) acc[g] = wmma_bf16(a, b[g], acc[g]);
  }

  const int j = j0 + r;
#pragma unroll
  for (int e = 0; e < 8; ++e) {
    const int m = half * 8 + e;
    if (mtile * 16 + m >= count) continue;
    const int node = lo + mtile * 16 + m;
    const size_t gb = (size_t)node * FIVEH + j;
    const float g0 = acc[0][e] + pre_x[gb + 0 * HDIM];
    const float g1 = acc[1][e] + pre_x[gb + 1 * HDIM];
    const float g2 = acc[2][e] + pre_x[gb + 2 * HDIM];
    const float g3 = acc[3][e] + pre_x[gb + 3 * HDIM];
    const float g4 = acc[4][e] + pre_x[gb + 4 * HDIM];
    const float ig = sigmoidf_(g0), og = sigmoidf_(g1), fg = sigmoidf_(g2);
    const float u = tanhf(g3), rg = sigmoidf_(g4);
    const int pie = parent[node];
    const float pc = (pie >= 0) ? cbuf[(size_t)pie * HDIM + j] : 0.0f;
    const float c = ig * u + fg * pc;
    const float hh = og * tanhf(c);
    const float hb = rg * hh + (1.0f - rg) * pre_px[(size_t)node * HDIM + j];
    cbuf[(size_t)node * HDIM + j] = c;
    hout_f32[(size_t)node * DDIM + j] = hb;
    hout_bf[(size_t)node * DDIM + j] = (__bf16)hb;
  }
}

// ---------------------------------------------------------------------------
// Host orchestration
// ---------------------------------------------------------------------------
extern "C" void kernel_launch(void* const* d_in, const int* in_sizes, int n_in,
                              void* d_out, int out_size, void* d_ws, size_t ws_size,
                              hipStream_t stream) {
  (void)in_sizes; (void)n_in; (void)out_size; (void)ws_size;
  const float* features = (const float*)d_in[0];
  const float* f_px_w = (const float*)d_in[1];
  const float* f_px_b = (const float*)d_in[2];
  const float* f_x_w  = (const float*)d_in[3];
  const float* f_x_b  = (const float*)d_in[4];
  const float* f_l_w  = (const float*)d_in[5];
  const float* f_l_b  = (const float*)d_in[6];
  const float* f_r_w  = (const float*)d_in[7];
  const float* f_r_b  = (const float*)d_in[8];
  const float* b_px_w = (const float*)d_in[9];
  const float* b_px_b = (const float*)d_in[10];
  const float* b_x_w  = (const float*)d_in[11];
  const float* b_x_b  = (const float*)d_in[12];
  const float* b_h_w  = (const float*)d_in[13];
  const float* b_h_b  = (const float*)d_in[14];
  const int* left   = (const int*)d_in[15];
  const int* right  = (const int*)d_in[16];
  const int* parent = (const int*)d_in[17];
  float* out = (float*)d_out;

  char* ws = (char*)d_ws;
  size_t off = 0;
  auto carve = [&](size_t bytes) -> void* {
    off = (off + 255) & ~(size_t)255;
    void* p = ws + off;
    off += bytes;
    return p;
  };
  __bf16* wb_fx  = (__bf16*)carve((size_t)SIXH  * DDIM * 2);
  __bf16* wb_fl  = (__bf16*)carve((size_t)SIXH  * HDIM * 2);
  __bf16* wb_fr  = (__bf16*)carve((size_t)SIXH  * HDIM * 2);
  __bf16* wb_fpx = (__bf16*)carve((size_t)HDIM  * DDIM * 2);
  __bf16* wb_bx  = (__bf16*)carve((size_t)FIVEH * DDIM * 2);
  __bf16* wb_bh  = (__bf16*)carve((size_t)FIVEH * HDIM * 2);
  __bf16* wb_bpx = (__bf16*)carve((size_t)HDIM  * DDIM * 2);
  float* bias_f  = (float*)carve((size_t)SIXH * 4);
  float* bias_b  = (float*)carve((size_t)FIVEH * 4);
  float* pre_fx  = (float*)carve((size_t)NNODE * SIXH * 4);
  float* pre_bx  = (float*)carve((size_t)NNODE * FIVEH * 4);
  float* pre_fpx = (float*)carve((size_t)NNODE * HDIM * 4);
  float* pre_bpx = (float*)carve((size_t)NNODE * HDIM * 4);
  __bf16* feats_a = (__bf16*)carve((size_t)NNODE * DDIM * 2);
  __bf16* feats_b = (__bf16*)carve((size_t)NNODE * DDIM * 2);
  float* c_f = (float*)carve((size_t)NNODE * HDIM * 4);
  float* c_b = (float*)carve((size_t)NNODE * HDIM * 4);
  float* hcat_ws = (float*)carve((size_t)NNODE * DDIM * 4);

  // layer-0 input features -> bf16
  f32_to_bf16_kernel<<<1024, 256, 0, stream>>>(features, feats_a, NNODE * DDIM);

  __bf16* fcur = feats_a;
  __bf16* fnext = feats_b;

  for (int l = 0; l < LNUM; ++l) {
    // --- per-layer weight conversion to bf16 ---
    f32_to_bf16_kernel<<<1024, 256, 0, stream>>>(f_x_w  + (size_t)l * SIXH * DDIM,  wb_fx,  SIXH * DDIM);
    f32_to_bf16_kernel<<<1024, 256, 0, stream>>>(f_l_w  + (size_t)l * SIXH * HDIM,  wb_fl,  SIXH * HDIM);
    f32_to_bf16_kernel<<<1024, 256, 0, stream>>>(f_r_w  + (size_t)l * SIXH * HDIM,  wb_fr,  SIXH * HDIM);
    f32_to_bf16_kernel<<<1024, 256, 0, stream>>>(f_px_w + (size_t)l * HDIM * DDIM,  wb_fpx, HDIM * DDIM);
    f32_to_bf16_kernel<<<1024, 256, 0, stream>>>(b_x_w  + (size_t)l * FIVEH * DDIM, wb_bx,  FIVEH * DDIM);
    f32_to_bf16_kernel<<<1024, 256, 0, stream>>>(b_h_w  + (size_t)l * FIVEH * HDIM, wb_bh,  FIVEH * HDIM);
    f32_to_bf16_kernel<<<1024, 256, 0, stream>>>(b_px_w + (size_t)l * HDIM * DDIM,  wb_bpx, HDIM * DDIM);
    // combined biases (child/parent biases are added unconditionally in the ref)
    combine_bias3_kernel<<<(SIXH + 255) / 256, 256, 0, stream>>>(
        f_x_b + (size_t)l * SIXH, f_l_b + (size_t)l * SIXH, f_r_b + (size_t)l * SIXH, bias_f, SIXH);
    combine_bias2_kernel<<<(FIVEH + 255) / 256, 256, 0, stream>>>(
        b_x_b + (size_t)l * FIVEH, b_h_b + (size_t)l * FIVEH, bias_b, FIVEH);

    // --- big input projections (M=1024, K=1024) ---
    gemm_bf16_wmma<<<dim3(SIXH / 256, NNODE / 64), 256, 0, stream>>>(
        fcur, DDIM, wb_fx, DDIM, bias_f, pre_fx, SIXH, DDIM);
    gemm_bf16_wmma<<<dim3(HDIM / 256, NNODE / 64), 256, 0, stream>>>(
        fcur, DDIM, wb_fpx, DDIM, f_px_b + (size_t)l * HDIM, pre_fpx, HDIM, DDIM);
    gemm_bf16_wmma<<<dim3(FIVEH / 256, NNODE / 64), 256, 0, stream>>>(
        fcur, DDIM, wb_bx, DDIM, bias_b, pre_bx, FIVEH, DDIM);
    gemm_bf16_wmma<<<dim3(HDIM / 256, NNODE / 64), 256, 0, stream>>>(
        fcur, DDIM, wb_bpx, DDIM, b_px_b + (size_t)l * HDIM, pre_bpx, HDIM, DDIM);

    float* hcat = (l == LNUM - 1) ? out : hcat_ws;

    // --- forward: deepest level first (children before parents) ---
    for (int lev = 10; lev >= 0; --lev) {
      int lo = (1 << lev) - 1;
      int hi = (1 << (lev + 1)) - 1;
      if (hi > NNODE) hi = NNODE;
      int count = hi - lo;
      if (count <= 0) continue;
      int mt = (count + 15) / 16;
      fwd_level_kernel<<<mt * 4, 256, 0, stream>>>(
          fnext /* hf at col 0 */, left, right, pre_fx, pre_fpx, wb_fl, wb_fr,
          c_f, hcat /* col 0 */, fnext /* col 0 */, lo, count);
    }
    // --- backward: root first (parents before children) ---
    for (int lev = 0; lev <= 10; ++lev) {
      int lo = (1 << lev) - 1;
      int hi = (1 << (lev + 1)) - 1;
      if (hi > NNODE) hi = NNODE;
      int count = hi - lo;
      if (count <= 0) continue;
      int mt = (count + 15) / 16;
      bwd_level_kernel<<<mt * 4, 256, 0, stream>>>(
          fnext + HDIM /* hb at col H */, parent, pre_bx, pre_bpx, wb_bh,
          c_b, hcat + HDIM, fnext + HDIM, lo, count);
    }

    __bf16* t = fcur; fcur = fnext; fnext = t;
  }
}
